// FAB_24146306138537
// MI455X (gfx1250) — compile-verified, matrix operand order checked by reading, not run on hardware
//
#include <hip/hip_runtime.h>
#include <hip/hip_bf16.h>
#include <math.h>

// ---------------- problem constants ----------------
#define CC    192
#define HEADS 6
#define WS    16
#define SHIFT 8
#define NTOK  256            // WS*WS
#define HD    32             // CC/HEADS
#define HID   768            // 4*CC
#define NW    256            // (H/WS)*(W/WS)
#define MTOT  65536          // H*W

typedef __attribute__((ext_vector_type(16))) _Float16 v16h;
typedef __attribute__((ext_vector_type(8)))  float    v8f;

// ---------------------------------------------------------------------------
// CDNA5 async global->LDS copy (16B per lane), tracked on ASYNCcnt.
// LDS destination is the 32-bit LDS byte offset (addrspace(3) pointer value).
// ---------------------------------------------------------------------------
__device__ __forceinline__ void async_cp16(void* lds_p, const void* gptr) {
  unsigned l = (unsigned)(size_t)(__attribute__((address_space(3))) void*)lds_p;
  asm volatile("global_load_async_to_lds_b128 %0, %1, off"
               :: "v"(l), "v"(gptr) : "memory");
}
__device__ __forceinline__ void wait_async0() {
  asm volatile("s_wait_asynccnt 0x0" ::: "memory");
}

// ---------------------------------------------------------------------------
// A/B fragment loader for v_wmma_f32_16x16x32_f16 (wave32).
// Source is a [16][32] (or [n][k]) patch of f16, row stride `ld` halves.
// Per ISA 16-bit A layout: lane m (0..15) / half h = lane>>4:
//   VGPR vg<4 : K = 2*vg + 8*h ; VGPR vg>=4: K = 16 + 2*(vg-4) + 8*h
// Same packing reused for the B operand with [n][k]-major storage
// (weightsT, K rows, Vt) so fragment reads are contiguous half-pairs.
// ---------------------------------------------------------------------------
__device__ inline v16h frag_ld(const _Float16* base, int ld, int lane) {
  const int m  = lane & 15;
  const int hh = lane >> 4;
  const _Float16* row = base + (size_t)m * ld;
  v16h r;
#pragma unroll
  for (int vg = 0; vg < 8; ++vg) {
    int k0 = ((vg < 4) ? (2 * vg) : (16 + 2 * (vg - 4))) + 8 * hh;
    r[2 * vg]     = row[k0];
    r[2 * vg + 1] = row[k0 + 1];
  }
  return r;
}

// =====================================================================
// Kernel 1: transpose + f16-convert all weight matrices: WT[n][k]=W[k][n]
// =====================================================================
__global__ __launch_bounds__(256) void cvt_weights(
    const float* __restrict__ qkv_w, const float* __restrict__ proj_w,
    const float* __restrict__ fc1_w, const float* __restrict__ fc2_w,
    _Float16* __restrict__ qkvT, _Float16* __restrict__ projT,
    _Float16* __restrict__ fc1T, _Float16* __restrict__ fc2T) {
  int idx = blockIdx.x * 256 + threadIdx.x;
  if (idx < 110592) {                       // qkv: K=192, N=576
    int n = idx / 192, k = idx % 192;
    qkvT[idx] = (_Float16)qkv_w[k * 576 + n];
  } else if (idx < 147456) {                // proj: 192x192
    int i = idx - 110592; int n = i / 192, k = i % 192;
    projT[i] = (_Float16)proj_w[k * 192 + n];
  } else if (idx < 294912) {                // fc1: K=192, N=768
    int i = idx - 147456; int n = i / 192, k = i % 192;
    fc1T[i] = (_Float16)fc1_w[k * 768 + n];
  } else if (idx < 442368) {                // fc2: K=768, N=192
    int i = idx - 294912; int n = i / 768, k = i % 768;
    fc2T[i] = (_Float16)fc2_w[k * 192 + n];
  }
}

// =====================================================================
// Kernel 2: relative-position bias gather: biasPre[head][q][k]
// =====================================================================
__global__ __launch_bounds__(256) void bias_pre_kernel(
    const int* __restrict__ rpi, const float* __restrict__ rpb,
    float* __restrict__ biasPre) {
  int idx  = blockIdx.x * 256 + threadIdx.x;  // < 6*65536
  int head = idx >> 16;
  int qk   = idx & 65535;
  biasPre[idx] = rpb[rpi[qk] * HEADS + head];
}

// =====================================================================
// Kernel 3/7: LayerNorm (one wave per token, 6 channels per lane).
// SHIFTMAP=true additionally applies roll(-8,-8) + window partition.
// =====================================================================
template <bool SHIFTMAP>
__global__ __launch_bounds__(256) void ln_kernel(
    const float* __restrict__ x, const float* __restrict__ g,
    const float* __restrict__ b, _Float16* __restrict__ out) {
  const int wid = threadIdx.x >> 5, lane = threadIdx.x & 31;
  const int t = blockIdx.x * 8 + wid;          // output token
  int oi = t;
  if (SHIFTMAP) {
    int wnd = t >> 8, tok = t & 255;
    int hs = (wnd >> 4) * WS + (tok >> 4);
    int ws = (wnd & 15) * WS + (tok & 15);
    oi = (((hs + SHIFT) & 255) << 8) | ((ws + SHIFT) & 255);
  }
  const float* row = x + (size_t)oi * CC;
  float v[6], s = 0.f;
#pragma unroll
  for (int j = 0; j < 6; ++j) { v[j] = row[j * 32 + lane]; s += v[j]; }
#pragma unroll
  for (int m = 16; m >= 1; m >>= 1) s += __shfl_xor(s, m, 32);
  float mean = s * (1.0f / CC), ss = 0.f;
#pragma unroll
  for (int j = 0; j < 6; ++j) { float d = v[j] - mean; ss += d * d; }
#pragma unroll
  for (int m = 16; m >= 1; m >>= 1) ss += __shfl_xor(ss, m, 32);
  float r = rsqrtf(ss * (1.0f / CC) + 1e-5f);
#pragma unroll
  for (int j = 0; j < 6; ++j) {
    int c = j * 32 + lane;
    out[(size_t)t * CC + c] = (_Float16)((v[j] - mean) * r * g[c] + b[c]);
  }
}

// =====================================================================
// WMMA GEMM  C = A[M,K] @ Bt[N,K]^T (+ epilogue), double-buffered
// async-to-LDS pipeline. Block tile 128x64, 8 waves, 2x2 WMMA per wave.
// MODE 0: QKV scatter; 1: proj(+reverse/roll/residual); 2: fc1+GELU;
// MODE 3: fc2+residual.
// =====================================================================
template <int MODE>
__global__ __launch_bounds__(256) void gemm_wmma(
    const _Float16* __restrict__ A, const _Float16* __restrict__ Bt,
    int M, int K, int N,
    const float* __restrict__ bias, const float* __restrict__ addsrc,
    float* __restrict__ fout, _Float16* __restrict__ hout,
    _Float16* __restrict__ qo, _Float16* __restrict__ ko,
    _Float16* __restrict__ vo, float qscale) {
  __shared__ _Float16 As[2][128 * 32];
  __shared__ _Float16 Bs[2][64 * 32];
  const int tid = threadIdx.x, lane = tid & 31, wid = tid >> 5;
  const int waveM = wid & 3, waveN = wid >> 2;
  const int m0 = blockIdx.y * 128, n0 = blockIdx.x * 64;

  v8f acc[2][2] = {};

  // async fill of one k-step tile pair into buffer `buf`
  auto issue = [&](int buf, int kt) {
#pragma unroll
    for (int r = 0; r < 2; ++r) {                // A: 128x32 halves
      int linear = tid + r * 256;
      int row = linear >> 1, seg = (linear & 1) * 16;
      async_cp16(&As[buf][row * 32 + seg],
                 &A[(size_t)(m0 + row) * K + kt + seg]);
    }
    if (tid < 128) {                             // B: 64x32 halves
      int row = tid >> 1, seg = (tid & 1) * 16;
      async_cp16(&Bs[buf][row * 32 + seg],
                 &Bt[(size_t)(n0 + row) * K + kt + seg]);
    }
  };

  const int nk = K >> 5;
  issue(0, 0);
  for (int it = 0; it < nk; ++it) {
    wait_async0();          // own wave's async fills of buf `it&1` complete
    __syncthreads();        // all waves' fills visible; prev compute done
    if (it + 1 < nk) issue((it + 1) & 1, (it + 1) * 32);
    const int buf = it & 1;
    v16h af[2], bf[2];
    af[0] = frag_ld(&As[buf][(waveM * 32 + 0) * 32], 32, lane);
    af[1] = frag_ld(&As[buf][(waveM * 32 + 16) * 32], 32, lane);
    bf[0] = frag_ld(&Bs[buf][(waveN * 32 + 0) * 32], 32, lane);
    bf[1] = frag_ld(&Bs[buf][(waveN * 32 + 16) * 32], 32, lane);
#pragma unroll
    for (int i = 0; i < 2; ++i)
#pragma unroll
      for (int j = 0; j < 2; ++j)
        acc[i][j] = __builtin_amdgcn_wmma_f32_16x16x32_f16(
            false, af[i], false, bf[j], (short)0, acc[i][j], false, false);
  }

  const int hh = lane >> 4, nc = lane & 15;
#pragma unroll
  for (int i = 0; i < 2; ++i)
#pragma unroll
    for (int j = 0; j < 2; ++j)
#pragma unroll
      for (int vg = 0; vg < 8; ++vg) {
        int gr = m0 + waveM * 32 + i * 16 + vg + 8 * hh;
        int gc = n0 + waveN * 32 + j * 16 + nc;
        float val = acc[i][j][vg] + bias[gc];
        if (MODE == 0) {
          int which = gc / CC, hcol = gc % CC;
          int head = hcol >> 5, hd = hcol & 31;
          int wnd = gr >> 8, tok = gr & 255;
          size_t bidx = ((size_t)(wnd * HEADS + head) * NTOK + tok) * HD + hd;
          if (which == 0)       qo[bidx] = (_Float16)(val * qscale);
          else if (which == 1)  ko[bidx] = (_Float16)val;
          else vo[((size_t)(wnd * HEADS + head) * HD + hd) * NTOK + tok] =
                   (_Float16)val;
        } else if (MODE == 1) {
          int wnd = gr >> 8, tok = gr & 255;
          int hs = (wnd >> 4) * WS + (tok >> 4);
          int ws = (wnd & 15) * WS + (tok & 15);
          int oi = (((hs + SHIFT) & 255) << 8) | ((ws + SHIFT) & 255);
          fout[(size_t)oi * CC + gc] = val + addsrc[(size_t)oi * CC + gc];
        } else if (MODE == 2) {
          float gl = 0.5f * val * (1.0f + erff(val * 0.70710678118654752f));
          hout[(size_t)gr * HID + gc] = (_Float16)gl;
        } else {
          fout[(size_t)gr * CC + gc] = val + addsrc[(size_t)gr * CC + gc];
        }
      }
}

// =====================================================================
// Flash-style windowed attention. Block = (window, head), 16 waves x
// 16 query rows. K [256x32] / Vt [32x256] staged via async-to-LDS.
// Online softmax; P converted C-layout -> A-layout via per-wave LDS.
// =====================================================================
__global__ __launch_bounds__(512) void attn_kernel(
    const _Float16* __restrict__ q, const _Float16* __restrict__ k,
    const _Float16* __restrict__ vt, const float* __restrict__ biasPre,
    const float* __restrict__ mask, _Float16* __restrict__ ao) {
  __shared__ _Float16 ksh[NTOK * HD];     // 16 KB
  __shared__ _Float16 vsh[HD * NTOK];     // 16 KB (Vt: [hd][key])
  __shared__ _Float16 psh[16][16 * 32];   // 16 KB per-wave P staging
  const int wnd = blockIdx.x, head = blockIdx.y;
  const int tid = threadIdx.x, lane = tid & 31, wid = tid >> 5;
  const size_t base = (size_t)(wnd * HEADS + head) * NTOK * HD;

  { // cooperative K / Vt fill: 1024 x 16B each, 512 threads, async path
    const _Float16* kg = k  + base;
    const _Float16* vg = vt + base;
    async_cp16(&ksh[tid * 8],          kg + tid * 8);
    async_cp16(&ksh[(tid + 512) * 8],  kg + (tid + 512) * 8);
    async_cp16(&vsh[tid * 8],          vg + tid * 8);
    async_cp16(&vsh[(tid + 512) * 8],  vg + (tid + 512) * 8);
  }
  wait_async0();
  __syncthreads();

  const int q0 = wid * 16;
  const v16h qf = frag_ld(q + base + (size_t)q0 * HD, HD, lane);
  v8f o0 = {}, o1 = {};
  float mrow[8], srow[8];
#pragma unroll
  for (int vg = 0; vg < 8; ++vg) { mrow[vg] = -1e30f; srow[vg] = 0.f; }
  const int hh = lane >> 4, nc = lane & 15;
  const float* bP = biasPre + (size_t)head * 65536;
  const float* mP = mask    + (size_t)wnd  * 65536;

  for (int ch = 0; ch < 8; ++ch) {        // 32 keys per chunk
    v16h kf0 = frag_ld(&ksh[(ch * 32 + 0)  * HD], HD, lane);
    v16h kf1 = frag_ld(&ksh[(ch * 32 + 16) * HD], HD, lane);
    v8f z0 = {}, z1 = {};
    v8f s0 = __builtin_amdgcn_wmma_f32_16x16x32_f16(false, qf, false, kf0,
                                                    (short)0, z0, false, false);
    v8f s1 = __builtin_amdgcn_wmma_f32_16x16x32_f16(false, qf, false, kf1,
                                                    (short)0, z1, false, false);
#pragma unroll
    for (int vg = 0; vg < 8; ++vg) {
      int row = q0 + vg + 8 * hh;
      int c0 = ch * 32 + nc, c1 = c0 + 16;
      float a0 = s0[vg] + bP[row * 256 + c0] + mP[row * 256 + c0];
      float a1 = s1[vg] + bP[row * 256 + c1] + mP[row * 256 + c1];
      float cm = fmaxf(a0, a1);
#pragma unroll
      for (int m = 8; m >= 1; m >>= 1) cm = fmaxf(cm, __shfl_xor(cm, m, 32));
      float mnew = fmaxf(mrow[vg], cm);
      float sc = __expf(mrow[vg] - mnew);
      float p0 = __expf(a0 - mnew), p1 = __expf(a1 - mnew);
      float ps = p0 + p1;
#pragma unroll
      for (int m = 8; m >= 1; m >>= 1) ps += __shfl_xor(ps, m, 32);
      srow[vg] = srow[vg] * sc + ps;
      mrow[vg] = mnew;
      o0[vg] *= sc; o1[vg] *= sc;
      psh[wid][(vg + 8 * hh) * 32 + nc]      = (_Float16)p0;
      psh[wid][(vg + 8 * hh) * 32 + nc + 16] = (_Float16)p1;
    }
    v16h pf  = frag_ld(&psh[wid][0], 32, lane);
    v16h vf0 = frag_ld(&vsh[0 * NTOK  + ch * 32], NTOK, lane);
    v16h vf1 = frag_ld(&vsh[16 * NTOK + ch * 32], NTOK, lane);
    o0 = __builtin_amdgcn_wmma_f32_16x16x32_f16(false, pf, false, vf0,
                                                (short)0, o0, false, false);
    o1 = __builtin_amdgcn_wmma_f32_16x16x32_f16(false, pf, false, vf1,
                                                (short)0, o1, false, false);
  }
#pragma unroll
  for (int vg = 0; vg < 8; ++vg) {
    int row = q0 + vg + 8 * hh;
    float inv = 1.0f / srow[vg];
    size_t ob = ((size_t)wnd * NTOK + row) * CC + head * HD;
    ao[ob + nc]      = (_Float16)(o0[vg] * inv);
    ao[ob + 16 + nc] = (_Float16)(o1[vg] * inv);
  }
}

// =====================================================================
// launcher
// =====================================================================
extern "C" void kernel_launch(void* const* d_in, const int* in_sizes, int n_in,
                              void* d_out, int out_size, void* d_ws,
                              size_t ws_size, hipStream_t stream) {
  const float* x       = (const float*)d_in[0];
  const int*   rpi     = (const int*)  d_in[1];
  const float* amask   = (const float*)d_in[2];
  const float* n1g     = (const float*)d_in[3];
  const float* n1b     = (const float*)d_in[4];
  const float* qkv_w   = (const float*)d_in[5];
  const float* qkv_b   = (const float*)d_in[6];
  const float* rpb     = (const float*)d_in[7];
  const float* proj_w  = (const float*)d_in[8];
  const float* proj_b  = (const float*)d_in[9];
  const float* n2g     = (const float*)d_in[10];
  const float* n2b     = (const float*)d_in[11];
  const float* fc1_w   = (const float*)d_in[12];
  const float* fc1_b   = (const float*)d_in[13];
  const float* fc2_w   = (const float*)d_in[14];
  const float* fc2_b   = (const float*)d_in[15];
  float* out = (float*)d_out;

  char* ws = (char*)d_ws;
  const size_t MB = 1u << 20;
  _Float16* qkvT   = (_Float16*)(ws + 0);            // 221184 B
  _Float16* projT  = (_Float16*)(ws + 221184);       //  73728 B
  _Float16* fc1T   = (_Float16*)(ws + 294912);       // 294912 B
  _Float16* fc2T   = (_Float16*)(ws + 589824);       // 294912 B -> 884736
  float*    biasPre= (float*)   (ws + 1 * MB);       // 1.5 MB
  _Float16* xw     = (_Float16*)(ws + 4 * MB);       // 24 MB (reused: att_out)
  _Float16* attOut = xw;
  _Float16* qbuf   = (_Float16*)(ws + 32 * MB);      // 24 MB (reused: ln2 out)
  _Float16* y16    = qbuf;
  _Float16* kbuf   = (_Float16*)(ws + 58 * MB);      // 24 MB
  _Float16* vbuf   = (_Float16*)(ws + 84 * MB);      // 24 MB
  float*    xr     = (float*)   (ws + 110 * MB);     // 48 MB
  _Float16* h16    = (_Float16*)(ws + 160 * MB);     // 96 MB

  const float qscale = 0.17677669529663689f;         // HD^-0.5

  cvt_weights<<<1728, 256, 0, stream>>>(qkv_w, proj_w, fc1_w, fc2_w,
                                        qkvT, projT, fc1T, fc2T);
  bias_pre_kernel<<<1536, 256, 0, stream>>>(rpi, rpb, biasPre);
  ln_kernel<true><<<8192, 256, 0, stream>>>(x, n1g, n1b, xw);
  gemm_wmma<0><<<dim3(576 / 64, MTOT / 128), 256, 0, stream>>>(
      xw, qkvT, MTOT, CC, 576, qkv_b, nullptr, nullptr, nullptr,
      qbuf, kbuf, vbuf, qscale);
  attn_kernel<<<dim3(NW, HEADS), 512, 0, stream>>>(qbuf, kbuf, vbuf, biasPre,
                                                   amask, attOut);
  gemm_wmma<1><<<dim3(CC / 64, MTOT / 128), 256, 0, stream>>>(
      attOut, projT, MTOT, CC, CC, proj_b, x, xr, nullptr,
      nullptr, nullptr, nullptr, 0.f);
  ln_kernel<false><<<8192, 256, 0, stream>>>(xr, n2g, n2b, y16);
  gemm_wmma<2><<<dim3(HID / 64, MTOT / 128), 256, 0, stream>>>(
      y16, fc1T, MTOT, CC, HID, fc1_b, nullptr, nullptr, h16,
      nullptr, nullptr, nullptr, 0.f);
  gemm_wmma<3><<<dim3(CC / 64, MTOT / 128), 256, 0, stream>>>(
      h16, fc2T, MTOT, HID, CC, fc2_b, xr, out, nullptr,
      nullptr, nullptr, nullptr, 0.f);
}
